// DMGatedGCNConv_86268713108267
// MI455X (gfx1250) — matrix operand order repaired and checked by reading, not compiled, not run on hardware
//
#include <hip/hip_runtime.h>
#include <hip/hip_bf16.h>

typedef __attribute__((ext_vector_type(16))) __bf16 v16bf;
typedef __attribute__((ext_vector_type(8)))  __bf16 v8bf;
typedef __attribute__((ext_vector_type(8)))  float  v8f;

static constexpr int CN = 128;   // channels
static constexpr int PN = 3;     // hops

// ---------------- softmax over P for d -> dw [P,C] ----------------
__global__ void softmax_dw_kernel(const float* __restrict__ d, float* __restrict__ dw) {
    int c = threadIdx.x;           // 128 threads
    float a0 = d[0 * CN + c], a1 = d[1 * CN + c], a2 = d[2 * CN + c];
    float m = fmaxf(a0, fmaxf(a1, a2));
    float e0 = __expf(a0 - m), e1 = __expf(a1 - m), e2 = __expf(a2 - m);
    float inv = 1.0f / (e0 + e1 + e2);
    dw[0 * CN + c] = e0 * inv;
    dw[1 * CN + c] = e1 * inv;
    dw[2 * CN + c] = e2 * inv;
}

// ---------------- fp32 -> bf16 elementwise ----------------
__global__ void cvt_bf16_kernel(const float* __restrict__ in, __bf16* __restrict__ out, int n) {
    for (int i = blockIdx.x * blockDim.x + threadIdx.x; i < n; i += gridDim.x * blockDim.x)
        out[i] = (__bf16)in[i];
}

// ---------------- combined skip weight + bias ----------------
// Wcomb[o,k] = sum_p dw[p,o] * Wskip[p,o,k]   (bf16)
// bias_comb[o] = hop_bias[o] + sum_p dw[p,o] * cbias[p,o]
__global__ void build_comb_kernel(const float* __restrict__ Wskip,
                                  const float* __restrict__ cbias,
                                  const float* __restrict__ hop_bias,
                                  const float* __restrict__ dw,
                                  __bf16* __restrict__ wcomb,
                                  float* __restrict__ bias_comb) {
    int o = blockIdx.x;      // 128 blocks
    int k = threadIdx.x;     // 128 threads
    float s = 0.0f;
    for (int p = 0; p < PN; ++p)
        s += dw[p * CN + o] * Wskip[((size_t)p * CN + o) * CN + k];
    wcomb[(size_t)o * CN + k] = (__bf16)s;
    if (k == 0) {
        float b = hop_bias[o];
        for (int p = 0; p < PN; ++p) b += dw[p * CN + o] * cbias[p * CN + o];
        bias_comb[o] = b;
    }
}

// ---------------- WMMA GEMM: Y[N,128] = Abf16[N,128] @ Bbf16[128,128]^T + bias ----------------
// B stored row-major [out_channel][in_channel] (torch Linear weight layout).
// One wave per 16-row tile; wave computes all 8 column tiles; K looped in 4 chunks of 32.
__global__ void gemm_bf16_kernel(const __bf16* __restrict__ A,
                                 const __bf16* __restrict__ B,
                                 const float* __restrict__ bias,
                                 float* __restrict__ Y, int nrows) {
    const int lane = threadIdx.x & 31;
    const int wave = threadIdx.x >> 5;
    const int rowTile = blockIdx.x * 8 + wave;
    if (rowTile * 16 >= nrows) return;          // wave-uniform exit, EXEC stays full
    const int row0 = rowTile * 16;
    const int half = lane >> 4;                  // which half-wave
    const int l    = lane & 15;

    v8f acc[8] = {};                             // 8 col tiles x 8 floats

    const __bf16* arow = A + (size_t)(row0 + l) * CN;
    for (int kc = 0; kc < 4; ++kc) {
        const int k0 = kc * 32;
        // A fragment (16x32 bf16): lane<16 holds K[k0..k0+7],[k0+16..k0+23]; lane>=16 shifted by 8
        v8bf alo = *(const v8bf*)(arow + k0 + half * 8);
        v8bf ahi = *(const v8bf*)(arow + k0 + 16 + half * 8);
        v16bf afrag;
        #pragma unroll
        for (int i = 0; i < 8; ++i) { afrag[i] = alo[i]; afrag[8 + i] = ahi[i]; }

        #pragma unroll
        for (int nt = 0; nt < 8; ++nt) {
            // B fragment (32x16): lane l of half h holds B[k0+h*8 .. +7] and B[k0+16+h*8 .. +7][nt*16+l]
            // = W[nt*16+l][k0+h*8 ..], W[nt*16+l][k0+16+h*8 ..] (same k order as A)
            const __bf16* brow = B + (size_t)(nt * 16 + l) * CN + k0;
            v8bf blo = *(const v8bf*)(brow + half * 8);
            v8bf bhi = *(const v8bf*)(brow + 16 + half * 8);
            v16bf bfrag;
            #pragma unroll
            for (int i = 0; i < 8; ++i) { bfrag[i] = blo[i]; bfrag[8 + i] = bhi[i]; }
            acc[nt] = __builtin_amdgcn_wmma_f32_16x16x32_bf16(
                false, afrag, false, bfrag, (short)0, acc[nt], false, false);
        }
    }

    // C/D layout: VGPR r -> row = row0 + half*8 + r, col = nt*16 + l
    #pragma unroll
    for (int nt = 0; nt < 8; ++nt) {
        const int col = nt * 16 + l;
        const float bc = bias[col];
        #pragma unroll
        for (int r = 0; r < 8; ++r) {
            Y[(size_t)(row0 + half * 8 + r) * CN + col] = acc[nt][r] + bc;
        }
    }
}

// ---------------- degree helpers ----------------
__global__ void zero_f32_kernel(float* __restrict__ p, int n) {
    int i = blockIdx.x * blockDim.x + threadIdx.x;
    if (i < n) p[i] = 0.0f;
}

__global__ void deg_count_kernel(const int* __restrict__ ei, float* __restrict__ deg, int E) {
    int e = blockIdx.x * blockDim.x + threadIdx.x;
    if (e < E) atomicAdd(&deg[ei[E + e]], 1.0f);   // col = target index
}

__global__ void dinv_kernel(float* __restrict__ deg, int n) {
    int i = blockIdx.x * blockDim.x + threadIdx.x;
    if (i < n) {
        float t = deg[i];
        deg[i] = (t > 0.0f) ? rsqrtf(t) : 0.0f;
    }
}

// ---------------- edge message: one wave per edge, float4 lanes cover 128 channels ----------------
__global__ void edge_msg_kernel(const int* __restrict__ ei, const float* __restrict__ ew,
                                const float* __restrict__ kbuf, const float* __restrict__ qbuf,
                                const float* __restrict__ vbuf, const float* __restrict__ dinv,
                                const float* __restrict__ dwrow, float* __restrict__ out, int E) {
    const int e = blockIdx.x * 8 + (threadIdx.x >> 5);
    if (e >= E) return;
    const int lane = threadIdx.x & 31;
    const int j = ei[e];          // row = source
    const int i = ei[E + e];      // col = target
    const float norm = dinv[i] * dinv[j] * ew[e];

    const int c = lane * 4;
    const float4 kv  = *(const float4*)(kbuf + (size_t)i * CN + c);
    const float4 qv  = *(const float4*)(qbuf + (size_t)j * CN + c);
    const float4 vv  = *(const float4*)(vbuf + (size_t)j * CN + c);
    const float4 dwv = *(const float4*)(dwrow + c);

    float4 m;
    m.x = (norm * dwv.x) * vv.x / (1.0f + __expf(-(kv.x + qv.x)));
    m.y = (norm * dwv.y) * vv.y / (1.0f + __expf(-(kv.y + qv.y)));
    m.z = (norm * dwv.z) * vv.z / (1.0f + __expf(-(kv.z + qv.z)));
    m.w = (norm * dwv.w) * vv.w / (1.0f + __expf(-(kv.w + qv.w)));

    float* o = out + (size_t)i * CN + c;
    atomicAdd(o + 0, m.x);
    atomicAdd(o + 1, m.y);
    atomicAdd(o + 2, m.z);
    atomicAdd(o + 3, m.w);
}

// ---------------- launcher ----------------
extern "C" void kernel_launch(void* const* d_in, const int* in_sizes, int n_in,
                              void* d_out, int out_size, void* d_ws, size_t ws_size,
                              hipStream_t stream) {
    const float* x        = (const float*)d_in[0];
    const int*   ei[3]    = {(const int*)d_in[1], (const int*)d_in[2], (const int*)d_in[3]};
    const float* ew[3]    = {(const float*)d_in[4], (const float*)d_in[5], (const float*)d_in[6]};
    const float* Wk       = (const float*)d_in[7];
    const float* bk       = (const float*)d_in[8];
    const float* Wq       = (const float*)d_in[9];
    const float* bq       = (const float*)d_in[10];
    const float* Wv       = (const float*)d_in[11];
    const float* bv       = (const float*)d_in[12];
    const float* Wskip    = (const float*)d_in[13];
    const float* cbias    = (const float*)d_in[14];
    const float* dmat     = (const float*)d_in[15];
    const float* hop_bias = (const float*)d_in[16];
    float* out = (float*)d_out;

    const int N = in_sizes[0] / CN;     // 50000
    const int E = in_sizes[4];          // 800000

    // workspace carve (256B aligned)
    char* ws = (char*)d_ws;
    size_t off = 0;
    auto carve = [&](size_t bytes) -> void* {
        void* p = ws + off;
        off = (off + bytes + 255) & ~(size_t)255;
        return p;
    };
    __bf16* xb        = (__bf16*)carve((size_t)N * CN * 2);
    __bf16* wkb       = (__bf16*)carve((size_t)PN * CN * CN * 2);
    __bf16* wqb       = (__bf16*)carve((size_t)PN * CN * CN * 2);
    __bf16* wvb       = (__bf16*)carve((size_t)PN * CN * CN * 2);
    __bf16* wcomb     = (__bf16*)carve((size_t)CN * CN * 2);
    float*  dw        = (float*)carve((size_t)PN * CN * 4);
    float*  bias_comb = (float*)carve((size_t)CN * 4);
    float*  kbuf      = (float*)carve((size_t)N * CN * 4);
    float*  qbuf      = (float*)carve((size_t)N * CN * 4);
    float*  vbuf      = (float*)carve((size_t)N * CN * 4);
    float*  deg       = (float*)carve((size_t)N * 4);

    // 1) softmax of d over P
    softmax_dw_kernel<<<1, CN, 0, stream>>>(dmat, dw);

    // 2) bf16 conversions
    cvt_bf16_kernel<<<4096, 256, 0, stream>>>(x, xb, N * CN);
    cvt_bf16_kernel<<<192, 256, 0, stream>>>(Wk, wkb, PN * CN * CN);
    cvt_bf16_kernel<<<192, 256, 0, stream>>>(Wq, wqb, PN * CN * CN);
    cvt_bf16_kernel<<<192, 256, 0, stream>>>(Wv, wvb, PN * CN * CN);

    // 3) combined skip weight/bias (needs dw)
    build_comb_kernel<<<CN, CN, 0, stream>>>(Wskip, cbias, hop_bias, dw, wcomb, bias_comb);

    // 4) initialize out with combined skip GEMM (bias folded in; no zero pass needed)
    const int rowTiles  = (N + 15) / 16;          // 3125
    const int gemmGrid  = (rowTiles + 7) / 8;     // 391
    gemm_bf16_kernel<<<gemmGrid, 256, 0, stream>>>(xb, wcomb, bias_comb, out, N);

    // 5) per hop: projections -> degree -> edge messages (atomic scatter into out)
    const int nBlk = (N + 255) / 256;
    const int eBlk = (E + 255) / 256;
    const int edgeGrid = (E + 7) / 8;             // one wave per edge
    for (int p = 0; p < PN; ++p) {
        gemm_bf16_kernel<<<gemmGrid, 256, 0, stream>>>(xb, wkb + (size_t)p * CN * CN, bk + p * CN, kbuf, N);
        gemm_bf16_kernel<<<gemmGrid, 256, 0, stream>>>(xb, wqb + (size_t)p * CN * CN, bq + p * CN, qbuf, N);
        gemm_bf16_kernel<<<gemmGrid, 256, 0, stream>>>(xb, wvb + (size_t)p * CN * CN, bv + p * CN, vbuf, N);

        zero_f32_kernel<<<nBlk, 256, 0, stream>>>(deg, N);
        deg_count_kernel<<<eBlk, 256, 0, stream>>>(ei[p], deg, E);
        dinv_kernel<<<nBlk, 256, 0, stream>>>(deg, N);

        edge_msg_kernel<<<edgeGrid, 256, 0, stream>>>(ei[p], ew[p], kbuf, qbuf, vbuf,
                                                      deg, dw + p * CN, out, E);
    }
}
